// MODEL_2439541424466
// MI455X (gfx1250) — matrix operand order repaired and checked by reading, not compiled, not run on hardware
//
#include <hip/hip_runtime.h>
#include <hip/hip_bf16.h>
#include <math.h>

// Problem dims (fixed by the reference)
#define BB 256
#define SS 512
#define BS (BB * SS)   // 131072 rows
#define MM 128         // memory slots
#define DKq 128        // key/query dim
#define DVv 256        // value dim
#define FCf 128        // final fc dim

typedef __attribute__((ext_vector_type(16))) __bf16 v16bf;
typedef __attribute__((ext_vector_type(8)))  __bf16 v8bf;
typedef __attribute__((ext_vector_type(8)))  float  v8f;

// ---- bf16 helpers (bit-level, RNE) ----
__device__ __forceinline__ unsigned short f2bf(float x) {
  unsigned int u = __float_as_uint(x);
  u += 0x7FFFu + ((u >> 16) & 1u);
  return (unsigned short)(u >> 16);
}
__device__ __forceinline__ float bf2f(unsigned short h) {
  return __uint_as_float(((unsigned int)h) << 16);
}

// ---- WMMA fragment loaders (CDNA5 wave32 layouts, ISA 7.12.2) ----
// A: 16x32 bf16. lane = m + 16*kh. elems 0..7 <- K = kh*8+0..7 ; elems 8..15 <- K = 16+kh*8+0..7
__device__ __forceinline__ v16bf load_afrag(const unsigned short* A, size_t row0, int ld, int k0, int lane) {
  int m = lane & 15, kh = lane >> 4;
  const __bf16* p = (const __bf16*)A + (row0 + (size_t)m) * (size_t)ld + k0 + kh * 8;
  v8bf lo = *(const v8bf*)(p);
  v8bf hi = *(const v8bf*)(p + 16);
  v16bf r;
#pragma unroll
  for (int i = 0; i < 8; i++) { r[i] = lo[i]; r[i + 8] = hi[i]; }
  return r;
}
// B: 32x16 bf16, B[k][n] = W[n][k] with W row-major [N,K]. lane = n + 16*kh, elems 0..7 <- K = kh*8+0..7 ; elems 8..15 <- K = 16+kh*8+0..7
__device__ __forceinline__ v16bf load_bfrag(const unsigned short* W, size_t col0, int ld, int k0, int lane) {
  int n = lane & 15, kb = (lane >> 4) * 8;
  const __bf16* p = (const __bf16*)W + (col0 + (size_t)n) * (size_t)ld + k0 + kb;
  v8bf lo = *(const v8bf*)(p);
  v8bf hi = *(const v8bf*)(p + 16);
  v16bf r;
#pragma unroll
  for (int i = 0; i < 8; i++) { r[i] = lo[i]; r[i + 8] = hi[i]; }
  return r;
}

#define WMMA_BF16(a, b, c) \
  __builtin_amdgcn_wmma_f32_16x16x32_bf16(false, (a), false, (b), (short)0, (c), false, false)

// ---- f32 -> bf16 converter ----
__global__ void k_convert(const float* __restrict__ src, unsigned short* __restrict__ dst, int n) {
  int i = blockIdx.x * blockDim.x + threadIdx.x;
  if (i < n) dst[i] = f2bf(src[i]);
}

// ---- embedding gather (padding row 0 already zero in tables) ----
__global__ void k_gather(const int* __restrict__ q, const int* __restrict__ qa,
                         const float* __restrict__ qtab, const float* __restrict__ qatab,
                         unsigned short* __restrict__ QEh, unsigned short* __restrict__ QAEh) {
  int i = blockIdx.x;           // row in [0, BS)
  int t = threadIdx.x;          // 0..383
  if (t < DKq) {
    int qi = q[i];
    QEh[(size_t)i * DKq + t] = f2bf(qtab[(size_t)qi * DKq + t]);
  } else {
    int d = t - DKq;
    int qai = qa[i];
    QAEh[(size_t)i * DVv + d] = f2bf(qatab[(size_t)qai * DVv + d]);
  }
}

// ---- attention weights fused: Wattn = softmax(QEh @ memory_key^T) ----
// block 256 = 8 waves, each wave one 16x16 tile; softmax done in LDS, only weights hit global.
__global__ void __launch_bounds__(256) k_attn(const unsigned short* __restrict__ QEh,
                                              const unsigned short* __restrict__ MKh,
                                              float* __restrict__ Wattn) {
  __shared__ float ps[16][MM];
  int lane = threadIdx.x & 31, wave = threadIdx.x >> 5;
  size_t row0 = (size_t)blockIdx.x * 16;
  size_t col0 = (size_t)wave * 16;     // 8 waves cover N=128
  v8f c = {};
#pragma unroll
  for (int k0 = 0; k0 < DKq; k0 += 32) {
    v16bf a = load_afrag(QEh, row0, DKq, k0, lane);
    v16bf b = load_bfrag(MKh, col0, DKq, k0, lane);
    c = WMMA_BF16(a, b, c);
  }
  int n = lane & 15, mb = (lane >> 4) * 8;
#pragma unroll
  for (int r = 0; r < 8; r++) ps[mb + r][col0 + n] = c[r];
  __syncthreads();
  // row softmax: 16 threads per row, strided cols, shuffle reductions within 16-lane groups
  int t = threadIdx.x;
  int r = t >> 4, c16 = t & 15;
  float vals[8];
  float mx = -3.4e38f;
#pragma unroll
  for (int j = 0; j < 8; j++) { vals[j] = ps[r][c16 + 16 * j]; mx = fmaxf(mx, vals[j]); }
#pragma unroll
  for (int off = 8; off >= 1; off >>= 1) mx = fmaxf(mx, __shfl_xor(mx, off, 16));
  float sum = 0.f;
#pragma unroll
  for (int j = 0; j < 8; j++) { vals[j] = expf(vals[j] - mx); sum += vals[j]; }
#pragma unroll
  for (int off = 8; off >= 1; off >>= 1) sum += __shfl_xor(sum, off, 16);
  float inv = 1.f / sum;
#pragma unroll
  for (int j = 0; j < 8; j++)
    Wattn[(row0 + r) * MM + c16 + 16 * j] = vals[j] * inv;
}

// ---- fused erase/add gates: E = sigmoid(QAEh@We^T+be), A = tanh(QAEh@Wa^T+ba) ----
// block 256 = 8 waves; wave handles 2 col tiles for BOTH weight matrices:
// per K-chunk: 1 A-frag load -> 4 B-frag loads -> 4 WMMAs (4x A reuse).
__global__ void __launch_bounds__(256) k_ea2(const unsigned short* __restrict__ QAEh,
                                             const unsigned short* __restrict__ WEh,
                                             const unsigned short* __restrict__ WAh,
                                             const float* __restrict__ be,
                                             const float* __restrict__ ba,
                                             unsigned short* __restrict__ Eo,
                                             unsigned short* __restrict__ Ao) {
  int lane = threadIdx.x & 31, wave = threadIdx.x >> 5;
  size_t row0 = (size_t)blockIdx.x * 16;
  v8f ce[2] = {{}, {}};
  v8f ca[2] = {{}, {}};
#pragma unroll
  for (int k0 = 0; k0 < DVv; k0 += 32) {
    v16bf a = load_afrag(QAEh, row0, DVv, k0, lane);
#pragma unroll
    for (int j = 0; j < 2; j++) {
      size_t col0 = (size_t)(wave * 2 + j) * 16;   // 8 waves * 2 tiles cover N=256
      v16bf bE = load_bfrag(WEh, col0, DVv, k0, lane);
      ce[j] = WMMA_BF16(a, bE, ce[j]);
      v16bf bA = load_bfrag(WAh, col0, DVv, k0, lane);
      ca[j] = WMMA_BF16(a, bA, ca[j]);
    }
  }
  int n = lane & 15, mb = (lane >> 4) * 8;
#pragma unroll
  for (int j = 0; j < 2; j++) {
    size_t col = (size_t)(wave * 2 + j) * 16 + n;
    float bcE = be[col], bcA = ba[col];
#pragma unroll
    for (int r = 0; r < 8; r++) {
      float xe = ce[j][r] + bcE;
      Eo[(row0 + mb + r) * DVv + col] = f2bf(1.f / (1.f + expf(-xe)));
      float xa = ca[j][r] + bcA;
      Ao[(row0 + mb + r) * DVv + col] = f2bf(tanhf(xa));
    }
  }
}

// ---- sequential memory scan: state pinned in LDS, software-pipelined global loads ----
// grid (256, 4), block 64: block owns mem[b][:, y*64 .. y*64+63] (128x64 f32 = 32KB LDS)
__global__ void __launch_bounds__(64) k_scan(const float* __restrict__ Wa,
                                             const unsigned short* __restrict__ E,
                                             const unsigned short* __restrict__ A,
                                             const float* __restrict__ mem0,
                                             unsigned short* __restrict__ READS) {
  __shared__ float smem[MM * 64];
  __shared__ float sw[MM];
  int b = blockIdx.x;
  int t = threadIdx.x;                 // 0..63, owns one value column
  int d = blockIdx.y * 64 + t;         // global value-dim column
  for (int m = 0; m < MM; m++) smem[m * 64 + t] = mem0[(size_t)m * DVv + d];
  // preload step 0 into registers
  size_t row = (size_t)b * SS;
  float w0 = Wa[row * MM + t];
  float w1 = Wa[row * MM + 64 + t];
  float e  = bf2f(E[row * DVv + d]);
  float a  = bf2f(A[row * DVv + d]);
  __syncthreads();
  for (int s = 0; s < SS; s++) {
    sw[t] = w0;
    sw[64 + t] = w1;
    __syncthreads();                   // sw for step s visible
    // issue next step's global loads early: latency overlaps the LDS update loop
    float w0n = 0.f, w1n = 0.f, en = 0.f, an = 0.f;
    if (s + 1 < SS) {
      w0n = Wa[(row + 1) * MM + t];
      w1n = Wa[(row + 1) * MM + 64 + t];
      en  = bf2f(E[(row + 1) * DVv + d]);
      an  = bf2f(A[(row + 1) * DVv + d]);
    }
    float acc = 0.f;
#pragma unroll 8
    for (int m = 0; m < MM; m++) {
      float w = sw[m];
      float mv = smem[m * 64 + t];
      mv = mv * (1.f - w * e) + w * a;
      smem[m * 64 + t] = mv;
      acc += w * mv;
    }
    READS[row * DVv + d] = f2bf(acc);
    __syncthreads();                   // all reads of sw done before next overwrite
    w0 = w0n; w1 = w1n; e = en; a = an;
    row++;
  }
}

// ---- final head: h = tanh([reads|qe] @ W_read^T + b_read); out = h @ W_pred + b_pred ----
// block 128 = 4 waves, each wave 2 col tiles (2x A reuse), then W_pred dot in LDS.
__global__ void __launch_bounds__(128) k_final(const unsigned short* __restrict__ READS,
                                               const unsigned short* __restrict__ QEh,
                                               const unsigned short* __restrict__ WRh,
                                               const float* __restrict__ b_read,
                                               const float* __restrict__ W_pred,
                                               const float* __restrict__ b_pred,
                                               float* __restrict__ out) {
  __shared__ float hs[16][FCf];
  int lane = threadIdx.x & 31, wave = threadIdx.x >> 5;  // wave 0..3
  size_t row0 = (size_t)blockIdx.x * 16;
  v8f c[2] = {{}, {}};
#pragma unroll
  for (int k0 = 0; k0 < DVv + FCf; k0 += 32) {
    v16bf a = (k0 < DVv) ? load_afrag(READS, row0, DVv, k0, lane)
                         : load_afrag(QEh, row0, DKq, k0 - DVv, lane);
#pragma unroll
    for (int j = 0; j < 2; j++) {
      size_t col0 = (size_t)(wave + 4 * j) * 16;         // 4 waves * 2 tiles cover FC=128
      v16bf b = load_bfrag(WRh, col0, DVv + FCf, k0, lane);
      c[j] = WMMA_BF16(a, b, c[j]);
    }
  }
  int n = lane & 15, mb = (lane >> 4) * 8;
#pragma unroll
  for (int j = 0; j < 2; j++) {
    size_t col0 = (size_t)(wave + 4 * j) * 16;
    float bcol = b_read[col0 + n];
#pragma unroll
    for (int r = 0; r < 8; r++)
      hs[mb + r][col0 + n] = tanhf(c[j][r] + bcol);
  }
  __syncthreads();
  int t = threadIdx.x;
  if (t < 16) {
    float acc = b_pred[0];
    for (int j = 0; j < FCf; j++) acc += hs[t][j] * W_pred[j];
    out[row0 + t] = acc;
  }
}

extern "C" void kernel_launch(void* const* d_in, const int* in_sizes, int n_in,
                              void* d_out, int out_size, void* d_ws, size_t ws_size,
                              hipStream_t stream) {
  (void)in_sizes; (void)n_in; (void)out_size; (void)ws_size;
  const int*   q_data  = (const int*)  d_in[0];
  const int*   qa_data = (const int*)  d_in[1];
  const float* qtab    = (const float*)d_in[2];
  const float* qatab   = (const float*)d_in[3];
  const float* memkey  = (const float*)d_in[4];
  const float* mem0    = (const float*)d_in[5];
  const float* W_erase = (const float*)d_in[6];
  const float* b_erase = (const float*)d_in[7];
  const float* W_add   = (const float*)d_in[8];
  const float* b_add   = (const float*)d_in[9];
  const float* W_read  = (const float*)d_in[10];
  const float* b_read  = (const float*)d_in[11];
  const float* W_pred  = (const float*)d_in[12];
  const float* b_pred  = (const float*)d_in[13];
  float* out = (float*)d_out;

  char* base = (char*)d_ws;
  size_t off = 0;
  auto alloc = [&](size_t bytes) -> void* {
    void* p = base + off;
    off = (off + bytes + 255) & ~(size_t)255;
    return p;
  };
  unsigned short* QEh   = (unsigned short*)alloc((size_t)BS * DKq * 2);         // 32 MB
  unsigned short* QAEh  = (unsigned short*)alloc((size_t)BS * DVv * 2);         // 64 MB
  float*          Wattn = (float*)         alloc((size_t)BS * MM * 4);          // 64 MB
  unsigned short* Ebuf  = (unsigned short*)alloc((size_t)BS * DVv * 2);         // 64 MB
  unsigned short* Abuf  = (unsigned short*)alloc((size_t)BS * DVv * 2);         // 64 MB
  unsigned short* READS = (unsigned short*)alloc((size_t)BS * DVv * 2);         // 64 MB
  unsigned short* MKh   = (unsigned short*)alloc((size_t)MM * DKq * 2);
  unsigned short* WEh   = (unsigned short*)alloc((size_t)DVv * DVv * 2);
  unsigned short* WAh   = (unsigned short*)alloc((size_t)DVv * DVv * 2);
  unsigned short* WRh   = (unsigned short*)alloc((size_t)FCf * (DVv + FCf) * 2);

  // 1) weights -> bf16
  k_convert<<<(MM * DKq + 255) / 256, 256, 0, stream>>>(memkey, MKh, MM * DKq);
  k_convert<<<(DVv * DVv + 255) / 256, 256, 0, stream>>>(W_erase, WEh, DVv * DVv);
  k_convert<<<(DVv * DVv + 255) / 256, 256, 0, stream>>>(W_add, WAh, DVv * DVv);
  k_convert<<<(FCf * (DVv + FCf) + 255) / 256, 256, 0, stream>>>(W_read, WRh, FCf * (DVv + FCf));

  // 2) embedding gathers -> bf16 activations
  k_gather<<<BS, DKq + DVv, 0, stream>>>(q_data, qa_data, qtab, qatab, QEh, QAEh);

  // 3) attention weights: fused GEMM + softmax (precomputed for all timesteps)
  k_attn<<<BS / 16, 256, 0, stream>>>(QEh, MKh, Wattn);

  // 4) fused erase/add gates (precomputed for all timesteps)
  k_ea2<<<BS / 16, 256, 0, stream>>>(QAEh, WEh, WAh, b_erase, b_add, Ebuf, Abuf);

  // 5) the only truly sequential part: LDS-resident, software-pipelined memory scan
  k_scan<<<dim3(BB, 4), 64, 0, stream>>>(Wattn, Ebuf, Abuf, mem0, READS);

  // 6) prediction head (fused two-source GEMM + tanh + W_pred reduction)
  k_final<<<BS / 16, 128, 0, stream>>>(READS, QEh, WRh, b_read, W_pred, b_pred, out);
}
